// InductiveGAT_51333449121926
// MI455X (gfx1250) — matrix operand order repaired and checked by reading, not compiled, not run on hardware
//
#include <hip/hip_runtime.h>
#include <hip/hip_bf16.h>
#include <math.h>

#define N_NODES 20000
#define N_EDGES 320000
#define NFEAT   50
#define ALPHA   0.2f
#define C_WIDTH 1024   // concat width layers 1-2 (4 heads * 256)

typedef __attribute__((ext_vector_type(16))) _Float16 v16h;
typedef __attribute__((ext_vector_type(8)))  _Float16 v8h;
typedef __attribute__((ext_vector_type(8)))  float    v8f;

union FragU { v16h v; v8h h2[2]; };

// ---------------------------------------------------------------- utilities
__device__ __forceinline__ float leakyf(float x) { return x > 0.f ? x : ALPHA * x; }
__device__ __forceinline__ float eluf(float x)   { return x > 0.f ? x : (__expf(x) - 1.f); }

__device__ __forceinline__ void atomicMaxFloat(float* addr, float v) {
    // int-max for non-negative, uint-min for negative: totally ordered vs both.
    if (v >= 0.f) atomicMax((int*)addr, __float_as_int(v));
    else          atomicMin((unsigned int*)addr, __float_as_uint(v));
}

__global__ void gat_fill(float* __restrict__ p, float v, int n) {
    int i = blockIdx.x * blockDim.x + threadIdx.x;
    if (i < n) p[i] = v;
}

// x (N,50) f32 -> xh (N,64) f16, zero padded K
__global__ void gat_pack_x(const float* __restrict__ x, _Float16* __restrict__ xh) {
    int i = blockIdx.x * blockDim.x + threadIdx.x;
    if (i >= N_NODES * 64) return;
    int n = i >> 6, k = i & 63;
    xh[i] = (k < NFEAT) ? (_Float16)x[n * NFEAT + k] : (_Float16)0.f;
}

// W (H,Kreal,Freal) f32 -> Wp packed in B-fragment order, zero padded.
// Wp[((h*kT+kt)*nT+nt)*512 + lane*16 + t] = W[h][kt*32 + t + (lane>>4)*16][nt*16 + (lane&15)]
__global__ void gat_pack_w(const float* __restrict__ W, _Float16* __restrict__ Wp,
                           int H, int Kreal, int Freal, int kTiles, int nTiles) {
    int tid = blockIdx.x * blockDim.x + threadIdx.x;
    int total = H * kTiles * nTiles * 512;
    if (tid >= total) return;
    int t    = tid & 15;
    int lane = (tid >> 4) & 31;
    int frag = tid >> 9;
    int nt = frag % nTiles; frag /= nTiles;
    int kt = frag % kTiles;
    int hd = frag / kTiles;
    int k = kt * 32 + ((lane >> 4) << 4) + t;
    int n = nt * 16 + (lane & 15);
    _Float16 val = (_Float16)0.f;
    if (k < Kreal && n < Freal)
        val = (_Float16)W[((size_t)hd * Kreal + k) * Freal + n];
    Wp[tid] = val;
}

// ------------------------------------------------------------------- GEMM
// Out[h][row][col] = sum_k A[row][k] * W[h][k][col], f16 inputs, f32 accum.
// grid: (M/16, nTiles/4, H); block: 128 (4 waves), wave w -> col tile blockIdx.y*4+w
__global__ void gat_wmma_gemm(const _Float16* __restrict__ A,
                              const _Float16* __restrict__ Bp,
                              float* __restrict__ Out,
                              int Kpad, int kTiles, int nTiles, int Fout) {
    const int lane = threadIdx.x & 31;
    const int wave = threadIdx.x >> 5;
    const int mt = blockIdx.x;
    const int nt = blockIdx.y * 4 + wave;
    const int hd = blockIdx.z;
    const int col = lane & 15;
    const int hi  = lane >> 4;

    const _Float16* Arow = A + (size_t)(mt * 16 + col) * Kpad + hi * 16; // col==m here
    const _Float16* Bhead = Bp + (((size_t)hd * kTiles) * nTiles + nt) * 512 + (lane << 4);

    v8f acc = {};
    for (int kt = 0; kt < kTiles; ++kt) {
        FragU a, b;
        a.h2[0] = *reinterpret_cast<const v8h*>(Arow + kt * 32);
        a.h2[1] = *reinterpret_cast<const v8h*>(Arow + kt * 32 + 8);
        const _Float16* bp = Bhead + (size_t)kt * nTiles * 512;
        b.h2[0] = *reinterpret_cast<const v8h*>(bp);
        b.h2[1] = *reinterpret_cast<const v8h*>(bp + 8);
        acc = __builtin_amdgcn_wmma_f32_16x16x32_f16(
            false, a.v, false, b.v, (short)0, acc, false, false);
    }
    float* o = Out + (size_t)hd * N_NODES * Fout
                   + (size_t)(mt * 16 + hi * 8) * Fout + nt * 16 + col;
#pragma unroll
    for (int r = 0; r < 8; ++r) o[(size_t)r * Fout] = acc[r];
}

// ------------------------------------------------- attention score vectors
// s_src[h,n] = Wh[h,n,:].a[h,0:F], s_dst[h,n] = Wh[h,n,:].a[h,F:2F]. Wave per (h,n).
__global__ void gat_scores(const float* __restrict__ Wh, const float* __restrict__ avec,
                           float* __restrict__ ssrc, float* __restrict__ sdst,
                           int H, int Fpad, int Freal) {
    int wid  = (blockIdx.x * blockDim.x + threadIdx.x) >> 5;
    int lane = threadIdx.x & 31;
    if (wid >= N_NODES * H) return;
    int hd = wid / N_NODES, n = wid % N_NODES;
    const float* w  = Wh + ((size_t)hd * N_NODES + n) * Fpad;
    const float* as = avec + (size_t)hd * (2 * Freal);
    const float* ad = as + Freal;
    float p1 = 0.f, p2 = 0.f;
    for (int f = lane; f < Freal; f += 32) {
        float v = w[f];
        p1 += v * as[f];
        p2 += v * ad[f];
    }
    for (int o = 16; o > 0; o >>= 1) {
        p1 += __shfl_xor(p1, o, 32);
        p2 += __shfl_xor(p2, o, 32);
    }
    if (lane == 0) { ssrc[wid] = p1; sdst[wid] = p2; }
}

// ----------------------------------------------------------- edge kernels
__global__ void gat_edge_max(const int* __restrict__ src, const int* __restrict__ dst,
                             const float* __restrict__ ssrc, const float* __restrict__ sdst,
                             float* __restrict__ emax, int H) {
    int tid = blockIdx.x * blockDim.x + threadIdx.x;
    if (tid >= N_EDGES * H) return;
    int hd = tid / N_EDGES, e = tid % N_EDGES;
    int s = src[e], d = dst[e];
    float v = leakyf(ssrc[hd * N_NODES + s] + sdst[hd * N_NODES + d]);
    atomicMaxFloat(&emax[hd * N_NODES + d], v);
}

__global__ void gat_edge_exp(const int* __restrict__ src, const int* __restrict__ dst,
                             const float* __restrict__ ssrc, const float* __restrict__ sdst,
                             const float* __restrict__ emax, float* __restrict__ eexp,
                             float* __restrict__ esum, int H) {
    int tid = blockIdx.x * blockDim.x + threadIdx.x;
    if (tid >= N_EDGES * H) return;
    int hd = tid / N_EDGES, e = tid % N_EDGES;
    int s = src[e], d = dst[e];
    float v  = leakyf(ssrc[hd * N_NODES + s] + sdst[hd * N_NODES + d]);
    float ex = __expf(v - emax[hd * N_NODES + d]);
    eexp[tid] = ex;
    atomicAdd(&esum[hd * N_NODES + d], ex);
}

// wave per (h,e): out[d*outStride + h*headStride + f] += Wh[h,src,f] * att
__global__ void gat_aggregate(const int* __restrict__ src, const int* __restrict__ dst,
                              const float* __restrict__ eexp, const float* __restrict__ esum,
                              const float* __restrict__ Wh, float* __restrict__ out,
                              int H, int Fpad, int outStride, int headStride) {
    int wid  = (blockIdx.x * blockDim.x + threadIdx.x) >> 5;
    int lane = threadIdx.x & 31;
    if (wid >= N_EDGES * H) return;
    int hd = wid / N_EDGES, e = wid % N_EDGES;
    int s = src[e], d = dst[e];
    float att = eexp[wid] / (esum[hd * N_NODES + d] + 1e-16f);
    const float* w = Wh + ((size_t)hd * N_NODES + s) * Fpad;
    float* o = out + (size_t)d * outStride + hd * headStride;
    for (int f = lane; f < Fpad; f += 32)
        atomicAdd(&o[f], w[f] * att);
}

// ----------------------------------------------------------- layer finish
__global__ void gat_finish1(const float* __restrict__ hpre, float* __restrict__ x1,
                            _Float16* __restrict__ x1h) {
    int i = blockIdx.x * blockDim.x + threadIdx.x;
    if (i >= N_NODES * C_WIDTH) return;
    float v = eluf(eluf(hpre[i]));   // per-head elu, then concat elu
    x1[i]  = v;
    x1h[i] = (_Float16)v;
}

__global__ void gat_finish2(const float* __restrict__ hpre, const float* __restrict__ x1,
                            _Float16* __restrict__ x2h) {
    int i = blockIdx.x * blockDim.x + threadIdx.x;
    if (i >= N_NODES * C_WIDTH) return;
    float v = eluf(eluf(hpre[i]) + x1[i]);   // skip connection
    x2h[i] = (_Float16)v;
}

__global__ void gat_finish3(const float* __restrict__ h3, float* __restrict__ out) {
    int i = blockIdx.x * blockDim.x + threadIdx.x;
    if (i >= N_NODES * 121) return;
    int n = i / 121, f = i % 121;
    float v = h3[(size_t)n * 128 + f] * (1.f / 6.f);  // mean over 6 heads
    out[i] = 1.f / (1.f + __expf(-v));
}

// ------------------------------------------------------------------ driver
static inline int cdiv(int a, int b) { return (a + b - 1) / b; }

extern "C" void kernel_launch(void* const* d_in, const int* in_sizes, int n_in,
                              void* d_out, int out_size, void* d_ws, size_t ws_size,
                              hipStream_t stream) {
    (void)in_sizes; (void)n_in; (void)out_size; (void)ws_size;
    const float* x  = (const float*)d_in[0];
    const int*   ei = (const int*)d_in[1];
    const int* src = ei;
    const int* dst = ei + N_EDGES;
    const float* W1 = (const float*)d_in[2];
    const float* a1 = (const float*)d_in[3];
    const float* W2 = (const float*)d_in[4];
    const float* a2 = (const float*)d_in[5];
    const float* W3 = (const float*)d_in[6];
    const float* a3 = (const float*)d_in[7];
    float* out = (float*)d_out;

    // ----- workspace carve-out (256B aligned)
    char* base = (char*)d_ws;
    size_t off = 0;
    auto alloc = [&](size_t bytes) -> char* {
        char* p = base + off;
        off = (off + bytes + 255) & ~(size_t)255;
        return p;
    };
    _Float16* xh  = (_Float16*)alloc((size_t)N_NODES * 64 * 2);
    _Float16* W1p = (_Float16*)alloc((size_t)4 * 2 * 16 * 512 * 2);
    _Float16* W2p = (_Float16*)alloc((size_t)4 * 32 * 16 * 512 * 2);
    _Float16* W3p = (_Float16*)alloc((size_t)6 * 32 * 8 * 512 * 2);
    float* Wh   = (float*)alloc((size_t)4 * N_NODES * 256 * 4);  // also fits 6*N*128
    float* ssrc = (float*)alloc((size_t)6 * N_NODES * 4);
    float* sdst = (float*)alloc((size_t)6 * N_NODES * 4);
    float* emax = (float*)alloc((size_t)6 * N_NODES * 4);
    float* esum = (float*)alloc((size_t)6 * N_NODES * 4);
    float* eexp = (float*)alloc((size_t)6 * N_EDGES * 4);
    float* hpre = (float*)alloc((size_t)N_NODES * C_WIDTH * 4);  // layer3 reuses N*128
    float* x1   = (float*)alloc((size_t)N_NODES * C_WIDTH * 4);
    _Float16* x1h = (_Float16*)alloc((size_t)N_NODES * C_WIDTH * 2);
    _Float16* x2h = (_Float16*)alloc((size_t)N_NODES * C_WIDTH * 2);

    const int B = 256;
    const float NEG_INF = -INFINITY;

    // ----- pack inputs to f16 / fragment order
    gat_pack_x<<<cdiv(N_NODES * 64, B), B, 0, stream>>>(x, xh);
    gat_pack_w<<<cdiv(4 * 2 * 16 * 512, B), B, 0, stream>>>(W1, W1p, 4, NFEAT, 256, 2, 16);
    gat_pack_w<<<cdiv(4 * 32 * 16 * 512, B), B, 0, stream>>>(W2, W2p, 4, 1024, 256, 32, 16);
    gat_pack_w<<<cdiv(6 * 32 * 8 * 512, B), B, 0, stream>>>(W3, W3p, 6, 1024, 121, 32, 8);

    // ================= layer 1 (4 heads, Fout=256, K=50->64) =================
    gat_wmma_gemm<<<dim3(N_NODES / 16, 4, 4), 128, 0, stream>>>(xh, W1p, Wh, 64, 2, 16, 256);
    gat_scores<<<cdiv(N_NODES * 4 * 32, B), B, 0, stream>>>(Wh, a1, ssrc, sdst, 4, 256, 256);
    gat_fill<<<cdiv(4 * N_NODES, B), B, 0, stream>>>(emax, NEG_INF, 4 * N_NODES);
    gat_fill<<<cdiv(4 * N_NODES, B), B, 0, stream>>>(esum, 0.f, 4 * N_NODES);
    gat_fill<<<cdiv(N_NODES * C_WIDTH, B), B, 0, stream>>>(hpre, 0.f, N_NODES * C_WIDTH);
    gat_edge_max<<<cdiv(N_EDGES * 4, B), B, 0, stream>>>(src, dst, ssrc, sdst, emax, 4);
    gat_edge_exp<<<cdiv(N_EDGES * 4, B), B, 0, stream>>>(src, dst, ssrc, sdst, emax, eexp, esum, 4);
    gat_aggregate<<<cdiv(N_EDGES * 4 * 32, B), B, 0, stream>>>(src, dst, eexp, esum, Wh, hpre,
                                                               4, 256, C_WIDTH, 256);
    gat_finish1<<<cdiv(N_NODES * C_WIDTH, B), B, 0, stream>>>(hpre, x1, x1h);

    // ================= layer 2 (4 heads, Fout=256, K=1024, skip) =============
    gat_wmma_gemm<<<dim3(N_NODES / 16, 4, 4), 128, 0, stream>>>(x1h, W2p, Wh, 1024, 32, 16, 256);
    gat_scores<<<cdiv(N_NODES * 4 * 32, B), B, 0, stream>>>(Wh, a2, ssrc, sdst, 4, 256, 256);
    gat_fill<<<cdiv(4 * N_NODES, B), B, 0, stream>>>(emax, NEG_INF, 4 * N_NODES);
    gat_fill<<<cdiv(4 * N_NODES, B), B, 0, stream>>>(esum, 0.f, 4 * N_NODES);
    gat_fill<<<cdiv(N_NODES * C_WIDTH, B), B, 0, stream>>>(hpre, 0.f, N_NODES * C_WIDTH);
    gat_edge_max<<<cdiv(N_EDGES * 4, B), B, 0, stream>>>(src, dst, ssrc, sdst, emax, 4);
    gat_edge_exp<<<cdiv(N_EDGES * 4, B), B, 0, stream>>>(src, dst, ssrc, sdst, emax, eexp, esum, 4);
    gat_aggregate<<<cdiv(N_EDGES * 4 * 32, B), B, 0, stream>>>(src, dst, eexp, esum, Wh, hpre,
                                                               4, 256, C_WIDTH, 256);
    gat_finish2<<<cdiv(N_NODES * C_WIDTH, B), B, 0, stream>>>(hpre, x1, x2h);

    // ================= layer 3 (6 heads, Fout=121->128, mean, sigmoid) =======
    gat_wmma_gemm<<<dim3(N_NODES / 16, 2, 6), 128, 0, stream>>>(x2h, W3p, Wh, 1024, 32, 8, 128);
    gat_scores<<<cdiv(N_NODES * 6 * 32, B), B, 0, stream>>>(Wh, a3, ssrc, sdst, 6, 128, 121);
    gat_fill<<<cdiv(6 * N_NODES, B), B, 0, stream>>>(emax, NEG_INF, 6 * N_NODES);
    gat_fill<<<cdiv(6 * N_NODES, B), B, 0, stream>>>(esum, 0.f, 6 * N_NODES);
    gat_fill<<<cdiv(N_NODES * 128, B), B, 0, stream>>>(hpre, 0.f, N_NODES * 128);
    gat_edge_max<<<cdiv(N_EDGES * 6, B), B, 0, stream>>>(src, dst, ssrc, sdst, emax, 6);
    gat_edge_exp<<<cdiv(N_EDGES * 6, B), B, 0, stream>>>(src, dst, ssrc, sdst, emax, eexp, esum, 6);
    gat_aggregate<<<cdiv(N_EDGES * 6 * 32, B), B, 0, stream>>>(src, dst, eexp, esum, Wh, hpre,
                                                               6, 128, 128, 0);
    gat_finish3<<<cdiv(N_NODES * 121, B), B, 0, stream>>>(hpre, out);
}
